// MyLSTM_77970836292041
// MI455X (gfx1250) — compile-verified, matrix operand order checked by reading, not run on hardware
//
#include <hip/hip_runtime.h>
#include <math.h>

#define B_ 128
#define T_ 512
#define F_ 128
#define H_ 512
#define G_ 2048  // 4*H

typedef __bf16 bf16_t;
typedef bf16_t v16bf __attribute__((ext_vector_type(16)));
typedef float  v8f   __attribute__((ext_vector_type(8)));
typedef unsigned int v8u __attribute__((ext_vector_type(8)));

union BfVec { v8u u; v16bf b; };

__device__ __forceinline__ unsigned short f2bf(float f) {
  unsigned int u = __float_as_uint(f);
  u += 0x7FFFu + ((u >> 16) & 1u);   // round-to-nearest-even
  return (unsigned short)(u >> 16);
}

__device__ __forceinline__ float sigmoidf_(float x) {
  return 1.0f / (1.0f + __expf(-x));
}

// Fast tanh: clamp so exp stays finite, then (e-1)/(e+1). ~6 VALU ops
// (one v_exp_f32 transcendental) vs the large OCML tanh.
__device__ __forceinline__ float tanhf_(float x) {
  float xc = fminf(fmaxf(x, -15.0f), 15.0f);
  float e  = __expf(2.0f * xc);
  return (e - 1.0f) / (e + 1.0f);
}

// Load 16 bf16 elements for this lane of a 16x32 WMMA A/B operand from a
// K-contiguous row (row-major, K innermost). Layout per CDNA5 ISA 7.12.2:
//   pairs j=0..3 -> K = 8*half + 2j, pairs j=4..7 -> K = 16 + 8*half + 2(j-4)
__device__ __forceinline__ v16bf load_pairs(const unsigned short* base, unsigned off, int half) {
  const unsigned int* q = (const unsigned int*)(base + off);
  BfVec v;
  int o = 4 * half;
  v.u[0] = q[o + 0];  v.u[1] = q[o + 1];  v.u[2] = q[o + 2];  v.u[3] = q[o + 3];
  v.u[4] = q[o + 8];  v.u[5] = q[o + 9];  v.u[6] = q[o + 10]; v.u[7] = q[o + 11];
  return v.b;
}

// ---------------- converters / packers ----------------

__global__ void cvt_bf16(const float* __restrict__ in, unsigned short* __restrict__ out, int n) {
  int i = blockIdx.x * blockDim.x + threadIdx.x;
  if (i < n) out[i] = f2bf(in[i]);
}

// in: [K,N] f32 row-major  ->  out: [N,K] bf16 row-major (K contiguous)
__global__ void packT(const float* __restrict__ in, unsigned short* __restrict__ out, int K, int N) {
  int idx = blockIdx.x * blockDim.x + threadIdx.x;
  if (idx < K * N) {
    int k = idx / N;
    int n = idx % N;
    out[(size_t)n * K + k] = f2bf(in[(size_t)k * N + n]);
  }
}

// ---------------- fused 2-layer LSTM ----------------
// Grid: 8 blocks (16 batch rows each), 512 threads = 16 waves.
// Wave w owns h-columns [32w, 32w+32): 8 WMMA tiles = {i,f,g,o} x {sub0,sub1}.

__global__ __launch_bounds__(512, 1)
void lstm2_fused(const unsigned short* __restrict__ xbf,   // [B,T,F] bf16
                 const unsigned short* __restrict__ W1T,   // [G,F]  bf16
                 const unsigned short* __restrict__ U1T,   // [G,H]  bf16
                 const float* __restrict__ b1,             // [G]
                 const unsigned short* __restrict__ W2T,   // [G,H]  bf16
                 const unsigned short* __restrict__ U2T,   // [G,H]  bf16
                 const float* __restrict__ b2,             // [G]
                 unsigned short* __restrict__ seqbf,       // [B,T,H] bf16 (layer-1 hidden states)
                 float* __restrict__ h2f)                  // [B,H] f32 (layer-2 final hidden)
{
  __shared__ unsigned short hls[16 * H_];   // current hidden state, bf16, 16 KB

  const int tid  = threadIdx.x;
  const int wave = tid >> 5;
  const int lane = tid & 31;
  const int lm   = lane & 15;   // row (A) / col (B,C) within 16-tile
  const int half = lane >> 4;
  const int b0   = blockIdx.x * 16;
  const int hc   = wave * 32;   // this wave's h-column base

  for (int i = tid; i < 16 * H_; i += blockDim.x) hls[i] = 0;

  // gate-column index for each of the 8 tiles this lane touches
  unsigned ncol[8];
  float bias1[8], bias2[8];
#pragma unroll
  for (int j = 0; j < 8; ++j) {
    int g = j >> 1, s = j & 1;
    unsigned n = (unsigned)(g * H_ + hc + s * 16 + lm);
    ncol[j]  = n;
    bias1[j] = b1[n];
    bias2[j] = b2[n];
  }

  v8f c[2];
#pragma unroll
  for (int s = 0; s < 2; ++s)
#pragma unroll
    for (int r = 0; r < 8; ++r) c[s][r] = 0.0f;

  __syncthreads();

  // ======== Layer 1: h0=c0=0, input = x ========
  for (int t = 0; t < T_; ++t) {
    v8f acc[8];
#pragma unroll
    for (int j = 0; j < 8; ++j) {
      float bv = bias1[j];
#pragma unroll
      for (int r = 0; r < 8; ++r) acc[j][r] = bv;
    }

    const unsigned short* xrow = xbf + ((size_t)(b0 + lm) * T_ + t) * F_;
    if (t + 1 < T_) __builtin_prefetch(xrow + F_, 0, 3);

    // z += x_t @ W1   (K = 128).  Keep rolled: full unrolling makes the weight
    // tiles t-invariant -> LICM hoists them across the t-loop -> scratch spills.
#pragma unroll 1
    for (int k0 = 0; k0 < F_; k0 += 32) {
      v16bf a = load_pairs(xrow, (unsigned)k0, half);
#pragma unroll
      for (int j = 0; j < 8; ++j) {
        v16bf bm = load_pairs(W1T, ncol[j] * (unsigned)F_ + (unsigned)k0, half);
        acc[j] = __builtin_amdgcn_wmma_f32_16x16x32_bf16(false, a, false, bm,
                                                         (short)0, acc[j], false, false);
      }
    }
    // z += h_{t-1} @ U1   (K = 512, A from LDS). Unroll 2: addresses still vary
    // with k0 (no hoist), but gives the scheduler two iterations to pipeline.
#pragma unroll 2
    for (int k0 = 0; k0 < H_; k0 += 32) {
      v16bf a = load_pairs(hls, (unsigned)(lm * H_ + k0), half);
#pragma unroll
      for (int j = 0; j < 8; ++j) {
        v16bf bm = load_pairs(U1T, ncol[j] * (unsigned)H_ + (unsigned)k0, half);
        acc[j] = __builtin_amdgcn_wmma_f32_16x16x32_bf16(false, a, false, bm,
                                                         (short)0, acc[j], false, false);
      }
    }
    __syncthreads();   // everyone done reading hls

    // gates -> c,h ; write h to LDS + seq
#pragma unroll
    for (int s = 0; s < 2; ++s) {
#pragma unroll
      for (int r = 0; r < 8; ++r) {
        float iv = sigmoidf_(acc[0 + s][r]);
        float fv = sigmoidf_(acc[2 + s][r]);
        float gv = tanhf_(acc[4 + s][r]);
        float ov = sigmoidf_(acc[6 + s][r]);
        float cv = fv * c[s][r] + iv * gv;
        c[s][r] = cv;
        float hv = ov * tanhf_(cv);
        int m   = r + 8 * half;
        int col = hc + s * 16 + lm;
        unsigned short hb = f2bf(hv);
        hls[m * H_ + col] = hb;
        seqbf[((size_t)(b0 + m) * T_ + t) * H_ + col] = hb;
      }
    }
    __syncthreads();   // hls updated for next step
  }

  // seq (global) written by all 16 waves must be visible to all waves' reads
  __threadfence();
  __syncthreads();

  // ======== Layer 2: initial state = (h1_final, c1_final) ========
  // hls already holds h1_final (bf16); c[] already holds c1_final.
  for (int t = 0; t < T_; ++t) {
    v8f acc[8];
#pragma unroll
    for (int j = 0; j < 8; ++j) {
      float bv = bias2[j];
#pragma unroll
      for (int r = 0; r < 8; ++r) acc[j][r] = bv;
    }

    const unsigned short* srow = seqbf + ((size_t)(b0 + lm) * T_ + t) * H_;
    if (t + 1 < T_) __builtin_prefetch(srow + H_, 0, 3);

    // z += seq_t @ W2   (K = 512)
#pragma unroll 2
    for (int k0 = 0; k0 < H_; k0 += 32) {
      v16bf a = load_pairs(srow, (unsigned)k0, half);
#pragma unroll
      for (int j = 0; j < 8; ++j) {
        v16bf bm = load_pairs(W2T, ncol[j] * (unsigned)H_ + (unsigned)k0, half);
        acc[j] = __builtin_amdgcn_wmma_f32_16x16x32_bf16(false, a, false, bm,
                                                         (short)0, acc[j], false, false);
      }
    }
    // z += h_{t-1} @ U2   (K = 512, A from LDS)
#pragma unroll 2
    for (int k0 = 0; k0 < H_; k0 += 32) {
      v16bf a = load_pairs(hls, (unsigned)(lm * H_ + k0), half);
#pragma unroll
      for (int j = 0; j < 8; ++j) {
        v16bf bm = load_pairs(U2T, ncol[j] * (unsigned)H_ + (unsigned)k0, half);
        acc[j] = __builtin_amdgcn_wmma_f32_16x16x32_bf16(false, a, false, bm,
                                                         (short)0, acc[j], false, false);
      }
    }
    __syncthreads();

    const bool last = (t == T_ - 1);
#pragma unroll
    for (int s = 0; s < 2; ++s) {
#pragma unroll
      for (int r = 0; r < 8; ++r) {
        float iv = sigmoidf_(acc[0 + s][r]);
        float fv = sigmoidf_(acc[2 + s][r]);
        float gv = tanhf_(acc[4 + s][r]);
        float ov = sigmoidf_(acc[6 + s][r]);
        float cv = fv * c[s][r] + iv * gv;
        c[s][r] = cv;
        float hv = ov * tanhf_(cv);
        int m   = r + 8 * half;
        int col = hc + s * 16 + lm;
        hls[m * H_ + col] = f2bf(hv);
        if (last) h2f[(size_t)(b0 + m) * H_ + col] = hv;
      }
    }
    __syncthreads();
  }
}

// ---------------- dense + softmax ----------------
// One wave per batch row. logits = h2 @ Wd + bd ; softmax over 10 classes.
__global__ void dense_softmax(const float* __restrict__ h,    // [B,H]
                              const float* __restrict__ Wd,   // [H,10]
                              const float* __restrict__ bd,   // [10]
                              float* __restrict__ out)        // [B,10]
{
  const int b = blockIdx.x;
  const int lane = threadIdx.x;
  float acc[10];
#pragma unroll
  for (int cc = 0; cc < 10; ++cc) acc[cc] = 0.0f;
  for (int k = lane; k < H_; k += 32) {
    float hv = h[(size_t)b * H_ + k];
#pragma unroll
    for (int cc = 0; cc < 10; ++cc) acc[cc] += hv * Wd[(size_t)k * 10 + cc];
  }
#pragma unroll
  for (int cc = 0; cc < 10; ++cc)
    for (int off = 16; off > 0; off >>= 1)
      acc[cc] += __shfl_down(acc[cc], off, 32);
  if (lane == 0) {
    float mx = -3.402823e38f;
#pragma unroll
    for (int cc = 0; cc < 10; ++cc) { acc[cc] += bd[cc]; mx = fmaxf(mx, acc[cc]); }
    float s = 0.0f;
    float e[10];
#pragma unroll
    for (int cc = 0; cc < 10; ++cc) { e[cc] = __expf(acc[cc] - mx); s += e[cc]; }
    float inv = 1.0f / s;
#pragma unroll
    for (int cc = 0; cc < 10; ++cc) out[(size_t)b * 10 + cc] = e[cc] * inv;
  }
}

// ---------------- launch ----------------

extern "C" void kernel_launch(void* const* d_in, const int* in_sizes, int n_in,
                              void* d_out, int out_size, void* d_ws, size_t ws_size,
                              hipStream_t stream) {
  const float* x  = (const float*)d_in[0];
  const float* W1 = (const float*)d_in[1];
  const float* U1 = (const float*)d_in[2];
  const float* b1 = (const float*)d_in[3];
  const float* W2 = (const float*)d_in[4];
  const float* U2 = (const float*)d_in[5];
  const float* b2 = (const float*)d_in[6];
  const float* Wd = (const float*)d_in[7];
  const float* bd = (const float*)d_in[8];
  float* out = (float*)d_out;

  char* ws = (char*)d_ws;
  size_t off = 0;
  auto carve = [&](size_t bytes) {
    char* p = ws + off;
    off += (bytes + 255) & ~(size_t)255;
    return p;
  };
  unsigned short* xbf  = (unsigned short*)carve((size_t)B_ * T_ * F_ * 2);  // 16 MB
  unsigned short* W1T  = (unsigned short*)carve((size_t)G_ * F_ * 2);       // 0.5 MB
  unsigned short* U1T  = (unsigned short*)carve((size_t)G_ * H_ * 2);       // 2 MB
  unsigned short* W2T  = (unsigned short*)carve((size_t)G_ * H_ * 2);       // 2 MB
  unsigned short* U2T  = (unsigned short*)carve((size_t)G_ * H_ * 2);       // 2 MB
  unsigned short* seqb = (unsigned short*)carve((size_t)B_ * T_ * H_ * 2);  // 64 MB
  float*          h2f  = (float*)carve((size_t)B_ * H_ * 4);                // 0.25 MB

  const int n_x = B_ * T_ * F_;
  cvt_bf16<<<(n_x + 255) / 256, 256, 0, stream>>>(x, xbf, n_x);
  packT<<<(F_ * G_ + 255) / 256, 256, 0, stream>>>(W1, W1T, F_, G_);
  packT<<<(H_ * G_ + 255) / 256, 256, 0, stream>>>(U1, U1T, H_, G_);
  packT<<<(H_ * G_ + 255) / 256, 256, 0, stream>>>(W2, W2T, H_, G_);
  packT<<<(H_ * G_ + 255) / 256, 256, 0, stream>>>(U2, U2T, H_, G_);

  lstm2_fused<<<B_ / 16, 512, 0, stream>>>(xbf, W1T, U1T, b1, W2T, U2T, b2, seqb, h2f);

  dense_softmax<<<B_, 32, 0, stream>>>(h2f, Wd, bd, out);
}
